// DeviceMOE_5291399708787
// MI455X (gfx1250) — matrix operand order repaired and checked by reading, not compiled, not run on hardware
//
#include <hip/hip_runtime.h>
#include <hip/hip_bf16.h>

#define HID 1024
#define ITR 2048
#define NE  8
#define KS  64                 // K elements per pipeline stage (2 WMMA k-chunks)

// LDS row pitch: 128 data bytes + 16 pad = 144 B. Bank start = 36*r mod 64:
// distinct for r=0..15, spacing >=4 -> conflict-free b128 reads; 16B aligned.
// TDM reproduces this layout with pad_enable (4 dwords after every 32 dwords).
#define RPITCH  144
#define RP_E    72             // pitch in bf16 elements

typedef __attribute__((ext_vector_type(16))) __bf16 v16bf;
typedef __attribute__((ext_vector_type(8)))  __bf16 v8bf;
typedef __attribute__((ext_vector_type(4)))  __bf16 v4bf;
typedef __attribute__((ext_vector_type(8)))  float  v8f;
typedef __attribute__((ext_vector_type(4)))  float  v4f;
typedef __attribute__((ext_vector_type(4)))  unsigned int v4u;
typedef __attribute__((ext_vector_type(8)))  int  v8i;
typedef __attribute__((ext_vector_type(4)))  int  v4i;

#if __has_builtin(__builtin_amdgcn_s_wait_asynccnt)
#define WAIT_ASYNC(n) __builtin_amdgcn_s_wait_asynccnt(n)
#else
#define WAIT_ASYNC(n) asm volatile("s_wait_asynccnt %0" :: "n"(n) : "memory")
#endif

#if __has_builtin(__builtin_amdgcn_s_wait_tensorcnt)
#define WAIT_TENSOR(n) __builtin_amdgcn_s_wait_tensorcnt(n)
#else
#define WAIT_TENSOR(n) asm volatile("s_wait_tensorcnt %0" :: "n"(n) : "memory")
#endif

#if __has_builtin(__builtin_amdgcn_tensor_load_to_lds)
#define HAVE_TDM 1
#else
#define HAVE_TDM 0
#endif

// Async 16B/lane copy global -> LDS (ASYNCcnt-tracked, no VGPR data).
// loff: wave-relative LDS byte offset (dynamic LDS base == 0, no static LDS here).
__device__ __forceinline__ void async_b128(unsigned int loff, const void* g) {
  asm volatile("global_load_async_to_lds_b128 %0, %1, off"
               :: "v"(loff), "v"(g) : "memory");
}

#if HAVE_TDM
// TDM: load a (rows x 128B) tile, row stride rowstride_dw dwords, into LDS at
// lds_off with 16B pad after every 128B row (-> RPITCH=144 rows). Tensor dims
// are set equal to the tile (tile start address is passed directly; no OOB).
// This toolchain's builtin is the 6-arg clang-23 form:
//   (uint32x4 g0, int32x8 g1, int32x4 g2, int32x4 g3, int32x8 extra, i32 cpol)
__device__ __forceinline__ void tdm_load_2d(unsigned int lds_off, const void* g,
                                            int rows, int rowstride_dw) {
  const unsigned long long ga = (unsigned long long)g;
  const v4u g0 = { 1u,                                   // count=1 (valid), user mode
                   lds_off,                              // lds_addr
                   (unsigned int)ga,                     // global_addr lo
                   (unsigned int)(ga >> 32) | (2u << 30) };  // addr hi | type=2
  const v8i g1 = { (int)((2u << 16)      // data_size = 4B
                 | (1u << 20)            // pad_enable
                 | (4u << 22)            // pad_interval: 32 dwords (128B)
                 | (3u << 25)),          // pad_amount: 4 dwords (16B)
                   (int)(32u << 16),                     // tensor_dim0 = 32 dw
                   (int)((unsigned)rows << 16),          // tensor_dim1 = rows
                   (int)(32u << 16),                     // tile_dim0 = 32 dw
                   rows,                                 // tile_dim1 (tile_dim2=0)
                   rowstride_dw,                         // tensor_dim0_stride
                   0, 0 };
  const v4i g2 = { 1, 1, 0, 0 };        // tensor_dim2=1, tensor_dim3=1, tile_dim3=0
  const v4i g3 = { 0, 1 << 16, 0, 0 };  // dim3_stride=0, tensor_dim4=1, tile_dim4=0
  const v8i gx = { 0, 0, 0, 0, 0, 0, 0, 0 };
  __builtin_amdgcn_tensor_load_to_lds(g0, g1, g2, g3, gx, 0);
}
#endif

__device__ __forceinline__ v4bf cvt4(v4f v) {
  v4bf r;
  r[0] = (__bf16)v[0]; r[1] = (__bf16)v[1];
  r[2] = (__bf16)v[2]; r[3] = (__bf16)v[3];
  return r;
}

__device__ __forceinline__ v16bf cat8(v8bf lo, v8bf hi) {
  return __builtin_shufflevector(lo, hi, 0,1,2,3,4,5,6,7,8,9,10,11,12,13,14,15);
}

// ---------------------------------------------------------------------------
// fp32 -> bf16 bulk conversion (one pass; staging becomes a pure copy and the
// L2-resident weight working set shrinks to ~100MB).
// ---------------------------------------------------------------------------
__global__ __launch_bounds__(256) void cvt_f32_bf16(
    const float* __restrict__ src, __bf16* __restrict__ dst, long n)
{
  long i = ((long)blockIdx.x * blockDim.x + threadIdx.x) * 4;
  if (i + 3 < n) {
    *(v4bf*)(dst + i) = cvt4(*(const v4f*)(src + i));
  } else {
    for (; i < n; ++i) dst[i] = (__bf16)src[i];
  }
}

// ---------------------------------------------------------------------------
// Router: one wave32 per token; softmax + top-2; builds per-expert token lists.
// ---------------------------------------------------------------------------
__global__ __launch_bounds__(128) void moe_router(
    const float* __restrict__ x, const float* __restrict__ rw,
    float* __restrict__ combine, int* __restrict__ counts,
    int* __restrict__ idxb, int T)
{
  const int lane = threadIdx.x & 31;
  const int t = blockIdx.x * 4 + (threadIdx.x >> 5);
  if (t >= T) return;

  float acc[NE];
  #pragma unroll
  for (int e = 0; e < NE; ++e) acc[e] = 0.f;

  for (int h = lane; h < HID; h += 32) {
    const float xv = x[(size_t)t * HID + h];
    #pragma unroll
    for (int e = 0; e < NE; ++e)
      acc[e] = fmaf(xv, rw[e * HID + h], acc[e]);
  }
  #pragma unroll
  for (int e = 0; e < NE; ++e) {
    #pragma unroll
    for (int off = 16; off > 0; off >>= 1)
      acc[e] += __shfl_xor(acc[e], off, 32);
  }

  if (lane == 0) {
    float mx = acc[0];
    #pragma unroll
    for (int e = 1; e < NE; ++e) mx = fmaxf(mx, acc[e]);
    float p[NE], s = 0.f;
    #pragma unroll
    for (int e = 0; e < NE; ++e) { p[e] = __expf(acc[e] - mx); s += p[e]; }
    const float inv = 1.f / s;

    int a1 = 0; float b1 = -1.f;
    #pragma unroll
    for (int e = 0; e < NE; ++e) if (p[e] > b1) { b1 = p[e]; a1 = e; }
    int a2 = -1; float b2 = -1.f;
    #pragma unroll
    for (int e = 0; e < NE; ++e) if (e != a1 && p[e] > b2) { b2 = p[e]; a2 = e; }

    #pragma unroll
    for (int e = 0; e < NE; ++e)
      combine[(size_t)t * NE + e] = (e == a1) ? b1 * inv : ((e == a2) ? b2 * inv : 0.f);

    int s1 = atomicAdd(&counts[a1], 1); idxb[a1 * T + s1] = t;
    int s2 = atomicAdd(&counts[a2], 1); idxb[a2 * T + s2] = t;
  }
}

// ---------------------------------------------------------------------------
// GEMM1 + SiLU-gate. Block: (expert, 32-token tile, 64-i tile), 8 waves (2x4).
// Double-buffered pipeline, KS=64/stage. A tile: 1 async b128 per wave
// (token gather). B gate/up tiles: one TDM descriptor each (waves 0/1).
// ---------------------------------------------------------------------------
#define S1_A  0
#define S1_BG (32 * RPITCH)
#define S1_BU (96 * RPITCH)
#define STG1  (160 * RPITCH)      // 23040 B per stage

__global__ __launch_bounds__(256) void moe_gemm1(
    const __bf16* __restrict__ xb, const __bf16* __restrict__ w1b,
    const int* __restrict__ counts, const int* __restrict__ idxb,
    __bf16* __restrict__ ybuf, int T)
{
  extern __shared__ __bf16 smem[];      // 2 * STG1 bytes
  const int e = blockIdx.z;
  const int cnt = counts[e];
  const int slot0 = blockIdx.y * 32;
  if (slot0 >= cnt) return;             // uniform early-exit
  const int n0 = blockIdx.x * 64;

  const int tid = threadIdx.x, lane = tid & 31, wid = tid >> 5;
  const int wm = wid >> 2, wn = wid & 3;
  const int sub = lane >> 3;                 // 0..3: row within 4-row group
  const unsigned int cb = (lane & 7) * 16;   // byte offset within 128B row chunk

#if HAVE_TDM
  // A: wave `wid` copies rows 4*wid..4*wid+3 (one b128 per lane)
  const char* ag; unsigned int al;
  {
    const int lr = wid * 4 + sub;
    int slot = slot0 + lr; if (slot >= cnt) slot = cnt - 1;   // safe addr, data unused
    const int tok = idxb[e * T + slot];
    ag = (const char*)(xb + (size_t)tok * HID) + cb;
    al = S1_A + lr * RPITCH + cb;
  }
  // B: TDM descriptors (uniform), 64 rows x 128B, row stride HID bf16 = 512 dw
  const char* bg_g = (const char*)(w1b + ((size_t)e * 2 * ITR + n0) * HID);
  const char* bu_g = (const char*)(w1b + ((size_t)e * 2 * ITR + ITR + n0) * HID);
#else
  const char* gbase[5];
  unsigned int lbase[5];
  #pragma unroll
  for (int j = 0; j < 5; ++j) {
    const int q = wid * 5 + j;               // 0..39
    if (q < 8) {
      const int lr = q * 4 + sub;
      int slot = slot0 + lr; if (slot >= cnt) slot = cnt - 1;
      const int tok = idxb[e * T + slot];
      gbase[j] = (const char*)(xb + (size_t)tok * HID) + cb;
      lbase[j] = S1_A + lr * RPITCH + cb;
    } else if (q < 24) {
      const int lr = (q - 8) * 4 + sub;
      gbase[j] = (const char*)(w1b + ((size_t)e * 2 * ITR + n0 + lr) * HID) + cb;
      lbase[j] = S1_BG + lr * RPITCH + cb;
    } else {
      const int lr = (q - 24) * 4 + sub;
      gbase[j] = (const char*)(w1b + ((size_t)e * 2 * ITR + ITR + n0 + lr) * HID) + cb;
      lbase[j] = S1_BU + lr * RPITCH + cb;
    }
  }
#endif

  v8f accg = {};
  v8f accu = {};
  const int r  = lane & 15;
  const int kA = (lane >> 4) * 8;            // A frag: chunks at kA, kA+16 (elems)
  const int kB = (lane >> 4) * 8;            // B frag: chunks at kB, kB+16 (elems)

  // prologue: stage 0 into buffer 0
#if HAVE_TDM
  async_b128(al, ag);
  if (wid == 0)      tdm_load_2d(S1_BG, bg_g, 64, 512);
  else if (wid == 1) tdm_load_2d(S1_BU, bu_g, 64, 512);
#else
  #pragma unroll
  for (int j = 0; j < 5; ++j) async_b128(lbase[j], gbase[j]);
#endif

  const int nst = HID / KS;                  // 16 stages
  for (int s = 0; s < nst; ++s) {
    const int b = s & 1;
    if (s + 1 < nst) {
      const size_t koff = (size_t)(s + 1) * KS * 2;  // bytes
      const unsigned int lo = (b ^ 1) * STG1;
#if HAVE_TDM
      async_b128(al + lo, ag + koff);
      if (wid == 0)      tdm_load_2d(S1_BG + lo, bg_g + koff, 64, 512);
      else if (wid == 1) tdm_load_2d(S1_BU + lo, bu_g + koff, 64, 512);
      WAIT_ASYNC(1);                         // stage s A drained, s+1 in flight
      if (wid < 2) WAIT_TENSOR(1);           // stage s B drained (in-order per wave)
#else
      #pragma unroll
      for (int j = 0; j < 5; ++j) async_b128(lbase[j] + lo, gbase[j] + koff);
      WAIT_ASYNC(5);
#endif
    } else {
      WAIT_ASYNC(0);
#if HAVE_TDM
      if (wid < 2) WAIT_TENSOR(0);
#endif
    }
    __syncthreads();

    const __bf16* sa = smem + (size_t)b * (STG1 / 2) + (S1_A  / 2) + (wm * 16 + r) * RP_E;
    const __bf16* sg = smem + (size_t)b * (STG1 / 2) + (S1_BG / 2) + (wn * 16 + r) * RP_E;
    const __bf16* su = smem + (size_t)b * (STG1 / 2) + (S1_BU / 2) + (wn * 16 + r) * RP_E;
    #pragma unroll
    for (int c = 0; c < 2; ++c) {            // two 32-wide k-chunks
      const v16bf a  = cat8(*(const v8bf*)(sa + c * 32 + kA),
                            *(const v8bf*)(sa + c * 32 + kA + 16));
      const v16bf bg = cat8(*(const v8bf*)(sg + c * 32 + kB),
                            *(const v8bf*)(sg + c * 32 + kB + 16));
      const v16bf bu = cat8(*(const v8bf*)(su + c * 32 + kB),
                            *(const v8bf*)(su + c * 32 + kB + 16));
      accg = __builtin_amdgcn_wmma_f32_16x16x32_bf16(false, a, false, bg,
                                                     (short)0, accg, false, false);
      accu = __builtin_amdgcn_wmma_f32_16x16x32_bf16(false, a, false, bu,
                                                     (short)0, accu, false, false);
    }
    __syncthreads();                         // done reading buf b before refill
  }

  // epilogue: y = up * silu(gate)
  const int mbase = (lane >> 4) * 8;
  const int n = lane & 15;
  #pragma unroll
  for (int v = 0; v < 8; ++v) {
    const int slot = slot0 + wm * 16 + mbase + v;
    if (slot < cnt) {
      const float g = accg[v], u = accu[v];
      const float yv = u * g * (1.f / (1.f + __expf(-g)));
      ybuf[(size_t)(e * T + slot) * ITR + (n0 + wn * 16 + n)] = (__bf16)yv;
    }
  }
}

// ---------------------------------------------------------------------------
// GEMM2 + combine. Block: (expert, 32-token tile, 64-h tile).
// A tile: 1 async b128 per wave. B tile: one TDM descriptor (wave 0).
// ---------------------------------------------------------------------------
#define S2_A 0
#define S2_B (32 * RPITCH)
#define STG2 (96 * RPITCH)        // 13824 B per stage

__global__ __launch_bounds__(256) void moe_gemm2(
    const __bf16* __restrict__ ybuf, const __bf16* __restrict__ w2b,
    const int* __restrict__ counts, const int* __restrict__ idxb,
    const float* __restrict__ combine, float* __restrict__ out, int T)
{
  extern __shared__ __bf16 smem[];      // 2 * STG2 bytes
  const int e = blockIdx.z;
  const int cnt = counts[e];
  const int slot0 = blockIdx.y * 32;
  if (slot0 >= cnt) return;
  const int n0 = blockIdx.x * 64;       // over H

  const int tid = threadIdx.x, lane = tid & 31, wid = tid >> 5;
  const int wm = wid >> 2, wn = wid & 3;
  const int sub = lane >> 3;
  const unsigned int cb = (lane & 7) * 16;

#if HAVE_TDM
  const char* ag; unsigned int al;
  {
    const int lr = wid * 4 + sub;
    int slot = slot0 + lr; if (slot >= cnt) slot = cnt - 1;
    ag = (const char*)(ybuf + (size_t)(e * T + slot) * ITR) + cb;
    al = S2_A + lr * RPITCH + cb;
  }
  // B: w2 rows, stride ITR bf16 = 1024 dw
  const char* b_g = (const char*)(w2b + ((size_t)e * HID + n0) * ITR);
#else
  const char* gbase[3];
  unsigned int lbase[3];
  #pragma unroll
  for (int j = 0; j < 3; ++j) {
    const int q = wid * 3 + j;               // 0..23
    if (q < 8) {
      const int lr = q * 4 + sub;
      int slot = slot0 + lr; if (slot >= cnt) slot = cnt - 1;
      gbase[j] = (const char*)(ybuf + (size_t)(e * T + slot) * ITR) + cb;
      lbase[j] = S2_A + lr * RPITCH + cb;
    } else {
      const int lr = (q - 8) * 4 + sub;
      gbase[j] = (const char*)(w2b + ((size_t)e * HID + n0 + lr) * ITR) + cb;
      lbase[j] = S2_B + lr * RPITCH + cb;
    }
  }
#endif

  v8f acc = {};
  const int r  = lane & 15;
  const int kA = (lane >> 4) * 8;
  const int kB = (lane >> 4) * 8;

#if HAVE_TDM
  async_b128(al, ag);
  if (wid == 0) tdm_load_2d(S2_B, b_g, 64, 1024);
#else
  #pragma unroll
  for (int j = 0; j < 3; ++j) async_b128(lbase[j], gbase[j]);
#endif

  const int nst = ITR / KS;                  // 32 stages
  for (int s = 0; s < nst; ++s) {
    const int b = s & 1;
    if (s + 1 < nst) {
      const size_t koff = (size_t)(s + 1) * KS * 2;
      const unsigned int lo = (b ^ 1) * STG2;
#if HAVE_TDM
      async_b128(al + lo, ag + koff);
      if (wid == 0) tdm_load_2d(S2_B + lo, b_g + koff, 64, 1024);
      WAIT_ASYNC(1);
      if (wid == 0) WAIT_TENSOR(1);
#else
      #pragma unroll
      for (int j = 0; j < 3; ++j) async_b128(lbase[j] + lo, gbase[j] + koff);
      WAIT_ASYNC(3);
#endif
    } else {
      WAIT_ASYNC(0);
#if HAVE_TDM
      if (wid == 0) WAIT_TENSOR(0);
#endif
    }
    __syncthreads();

    const __bf16* sa = smem + (size_t)b * (STG2 / 2) + (S2_A / 2) + (wm * 16 + r) * RP_E;
    const __bf16* sb = smem + (size_t)b * (STG2 / 2) + (S2_B / 2) + (wn * 16 + r) * RP_E;
    #pragma unroll
    for (int c = 0; c < 2; ++c) {
      const v16bf a = cat8(*(const v8bf*)(sa + c * 32 + kA),
                           *(const v8bf*)(sa + c * 32 + kA + 16));
      const v16bf bb = cat8(*(const v8bf*)(sb + c * 32 + kB),
                            *(const v8bf*)(sb + c * 32 + kB + 16));
      acc = __builtin_amdgcn_wmma_f32_16x16x32_bf16(false, a, false, bb,
                                                    (short)0, acc, false, false);
    }
    __syncthreads();
  }

  const int mbase = (lane >> 4) * 8;
  const int n = lane & 15;
  #pragma unroll
  for (int v = 0; v < 8; ++v) {
    const int slot = slot0 + wm * 16 + mbase + v;
    if (slot < cnt) {
      const int t = idxb[e * T + slot];
      const float cw = combine[(size_t)t * NE + e];
      unsafeAtomicAdd(out + (size_t)t * HID + (n0 + wn * 16 + n), acc[v] * cw);
    }
  }
}

// ---------------------------------------------------------------------------
extern "C" void kernel_launch(void* const* d_in, const int* in_sizes, int n_in,
                              void* d_out, int out_size, void* d_ws, size_t ws_size,
                              hipStream_t stream) {
  const float* x  = (const float*)d_in[0];   // [1, T, H]
  const float* w1 = (const float*)d_in[1];   // [E, 2I, H]
  const float* w2 = (const float*)d_in[2];   // [E, H, I]
  const float* rw = (const float*)d_in[3];   // [E, H]
  float* out = (float*)d_out;

  const int T = in_sizes[0] / HID;
  const long n_x  = (long)T * HID;
  const long n_w1 = (long)NE * 2 * ITR * HID;
  const long n_w2 = (long)NE * HID * ITR;

  // workspace layout (all chunks 16B-aligned)
  char* ws = (char*)d_ws;
  size_t off = 0;
  float* combine = (float*)(ws + off); off += (size_t)T * NE * sizeof(float);
  int*   counts  = (int*)(ws + off);   off += 256;
  int*   idxb    = (int*)(ws + off);   off += (size_t)NE * T * sizeof(int);
  __bf16* xb     = (__bf16*)(ws + off); off += (size_t)n_x * 2;
  __bf16* w1b    = (__bf16*)(ws + off); off += (size_t)n_w1 * 2;
  __bf16* w2b    = (__bf16*)(ws + off); off += (size_t)n_w2 * 2;
  __bf16* ybuf   = (__bf16*)(ws + off);

  (void)hipMemsetAsync(counts, 0, NE * sizeof(int), stream);
  (void)hipMemsetAsync(out, 0, (size_t)out_size * sizeof(float), stream);

  cvt_f32_bf16<<<dim3((unsigned)((n_x  / 4 + 255) / 256)), dim3(256), 0, stream>>>(x,  xb,  n_x);
  cvt_f32_bf16<<<dim3((unsigned)((n_w1 / 4 + 255) / 256)), dim3(256), 0, stream>>>(w1, w1b, n_w1);
  cvt_f32_bf16<<<dim3((unsigned)((n_w2 / 4 + 255) / 256)), dim3(256), 0, stream>>>(w2, w2b, n_w2);

  moe_router<<<dim3((T + 3) / 4), dim3(128), 0, stream>>>(
      x, rw, combine, counts, idxb, T);

  moe_gemm1<<<dim3(ITR / 64, (T + 31) / 32, NE), dim3(256), 2 * STG1, stream>>>(
      xb, w1b, counts, idxb, ybuf, T);

  moe_gemm2<<<dim3(HID / 64, (T + 31) / 32, NE), dim3(256), 2 * STG2, stream>>>(
      ybuf, w2b, counts, idxb, combine, out, T);
}
